// MorphTagModel_39513699123784
// MI455X (gfx1250) — matrix operand order repaired and checked by reading, not compiled, not run on hardware
//
#include <hip/hip_runtime.h>
#include <hip/hip_bf16.h>
#include <math.h>

// ---------------------------------------------------------------------------
// MorphTagModel for MI455X (gfx1250, wave32, WMMA).
// GEMMs: bf16 x bf16 -> f32 via v_wmma_f32_16x16x32_bf16, fragments fed by
// ds_load_b128; double-buffered LDS + register prefetch; 512 threads/block,
// block tile 128x256, wave tile 32x64 (8 wmma per 12 ds_load_b128).
// ---------------------------------------------------------------------------

typedef __attribute__((ext_vector_type(16))) __bf16 v16bf;
typedef __attribute__((ext_vector_type(8)))  float  v8f;

union Frag16 {
  v16bf v;
  uint4 q[2];
};

#define D_DIM   768
#define N_TOK   8192
#define T_SEG   1024
#define F_LEN   16
#define L_LEN   8
#define CV_DIM  512
#define TV_DIM  512

#define BM 128
#define BN 256
#define BK 32
#define SA_STRIDE 40   // 80 B rows: 16B-aligned, bank-staggered
#define SB_STRIDE 40

// ---------------------------------------------------------------------------
// C[M,N] = op( A0@B0t^T + A1@B1t^T + bias ), A* bf16 [M,K*], B*t bf16 [N,K*].
// Block 512 threads = 16 waves (4x4); block tile 128x256; wave tile 32x64.
// Double-buffered LDS, one barrier per K-step.
// ---------------------------------------------------------------------------
__launch_bounds__(512, 1)
__global__ void gemm_bf16_kernel(const __bf16* __restrict__ A0,
                                 const __bf16* __restrict__ B0t, int K0,
                                 const __bf16* __restrict__ A1,
                                 const __bf16* __restrict__ B1t, int K1,
                                 const float* __restrict__ bias, int do_tanh,
                                 float* __restrict__ C, int M, int N) {
  __shared__ __align__(16) __bf16 sA[2][BM][SA_STRIDE];
  __shared__ __align__(16) __bf16 sBt[2][BN][SB_STRIDE];

  const int tid  = threadIdx.x;
  const int lane = tid & 31;
  const int wave = tid >> 5;
  const int wm   = wave >> 2;      // 0..3 : 32-row group
  const int wn   = wave & 3;       // 0..3 : 64-col group
  const int m16  = lane & 15;
  const int half = lane >> 4;

  const int gm0 = blockIdx.y * BM;
  const int gn0 = blockIdx.x * BN;

  // Per-thread staging coordinates (A: 1 uint4, B: 2 uint4 per thread).
  const int sa_m  = tid >> 2;            // 0..127
  const int sa_kq = (tid & 3) * 8;       // 0,8,16,24
  const int sb_n  = tid >> 1;            // 0..255
  const int sb_kq = (tid & 1) * 16;      // 0,16

  v8f acc[2][4];
#pragma unroll
  for (int i = 0; i < 2; ++i)
#pragma unroll
    for (int j = 0; j < 4; ++j) acc[i][j] = (v8f){};

  for (int seg = 0; seg < 2; ++seg) {
    const __bf16* A  = seg ? A1 : A0;
    const __bf16* Bt = seg ? B1t : B0t;
    const int     K  = seg ? K1 : K0;
    if (K == 0) continue;
    const int steps = K / BK;

    const __bf16* arow = A + (size_t)(gm0 + sa_m) * K + sa_kq;
    const __bf16* brow = Bt + (size_t)(gn0 + sb_n) * K + sb_kq;

    // Preload step 0 into registers, stage into buffer 0.
    uint4 ra  = *(const uint4*)arow;
    uint4 rb0 = ((const uint4*)brow)[0];
    uint4 rb1 = ((const uint4*)brow)[1];
    *(uint4*)&sA[0][sa_m][sa_kq]      = ra;
    *(uint4*)&sBt[0][sb_n][sb_kq]     = rb0;
    *(uint4*)&sBt[0][sb_n][sb_kq + 8] = rb1;
    __syncthreads();

    for (int ks = 0; ks < steps; ++ks) {
      const int cur = ks & 1;

      // Prefetch next tile into registers (overlaps with WMMA below).
      const bool more = (ks + 1) < steps;
      if (more) {
        const __bf16* an = arow + (size_t)(ks + 1) * BK;
        const __bf16* bn = brow + (size_t)(ks + 1) * BK;
        ra  = *(const uint4*)an;
        rb0 = ((const uint4*)bn)[0];
        rb1 = ((const uint4*)bn)[1];
      }

      // A fragments stay resident; B fragments streamed one at a time.
      Frag16 a[2];
#pragma unroll
      for (int i = 0; i < 2; ++i) {
        const __bf16* r = &sA[cur][wm * 32 + i * 16 + m16][8 * half];
        a[i].q[0] = *(const uint4*)r;
        a[i].q[1] = *(const uint4*)(r + 16);
      }
#pragma unroll
      for (int j = 0; j < 4; ++j) {
        Frag16 b;
        const __bf16* r = &sBt[cur][wn * 64 + j * 16 + m16][8 * half];
        b.q[0] = *(const uint4*)r;
        b.q[1] = *(const uint4*)(r + 16);
        acc[0][j] = __builtin_amdgcn_wmma_f32_16x16x32_bf16(
            false, a[0].v, false, b.v, (short)0, acc[0][j], false, false);
        acc[1][j] = __builtin_amdgcn_wmma_f32_16x16x32_bf16(
            false, a[1].v, false, b.v, (short)0, acc[1][j], false, false);
      }

      // Stage next tile into the other buffer; one barrier per step.
      if (more) {
        const int nxt = cur ^ 1;
        *(uint4*)&sA[nxt][sa_m][sa_kq]      = ra;
        *(uint4*)&sBt[nxt][sb_n][sb_kq]     = rb0;
        *(uint4*)&sBt[nxt][sb_n][sb_kq + 8] = rb1;
        __syncthreads();
      }
    }
  }

  // ---- epilogue: acc VGPR r -> row r + 8*half, col = lane&15 ----
#pragma unroll
  for (int i = 0; i < 2; ++i) {
#pragma unroll
    for (int r = 0; r < 8; ++r) {
      const int row = gm0 + wm * 32 + i * 16 + r + 8 * half;
#pragma unroll
      for (int j = 0; j < 4; ++j) {
        const int col = gn0 + wn * 64 + j * 16 + m16;
        float v = acc[i][j][r];
        if (bias) v += bias[col];
        if (do_tanh) v = tanhf(v);
        C[(size_t)row * N + col] = v;
      }
    }
  }
}

// ---------------------------------------------------------------------------
// Wt[n][k] = bf16(W[k][n])   (one-time per weight; writes coalesced along k)
// ---------------------------------------------------------------------------
__global__ void transpose_bf16_kernel(const float* __restrict__ W,
                                      __bf16* __restrict__ Wt, int K, int N) {
  const int idx = blockIdx.x * blockDim.x + threadIdx.x;  // over [N][K]
  const int n = idx / K;
  const int k = idx % K;
  Wt[idx] = (__bf16)W[(size_t)k * N + n];
}

// ---------------------------------------------------------------------------
// A_enc[m][k] = bf16(xemb[xtokens[m]][k])
// ---------------------------------------------------------------------------
__global__ void enc_gather_kernel(const float* __restrict__ xemb,
                                  const int* __restrict__ xtokens,
                                  __bf16* __restrict__ A_enc) {
  const int m = blockIdx.x;
  const float* src = xemb + (size_t)xtokens[m] * D_DIM;
  __bf16* dst = A_enc + (size_t)m * D_DIM;
  for (int d = threadIdx.x; d < D_DIM; d += blockDim.x) dst[d] = (__bf16)src[d];
}

// ---------------------------------------------------------------------------
// Segment sums via atomics.
// ---------------------------------------------------------------------------
__global__ void segsum_kernel(const int* __restrict__ tokens,
                              const float* __restrict__ token_ctx,
                              const float* __restrict__ char_emb,
                              float* __restrict__ ts, float* __restrict__ cs,
                              float* __restrict__ cnt) {
  const int i   = blockIdx.x;
  const int seg = tokens[3 * i + 0] - 1;
  const int idx = tokens[3 * i + 1];
  const int ch  = tokens[3 * i + 2];
  const float* tsrc = token_ctx + (size_t)idx * D_DIM;
  const float* csrc = char_emb + (size_t)ch * D_DIM;
  float* tdst = ts + (size_t)seg * D_DIM;
  float* cdst = cs + (size_t)seg * D_DIM;
  for (int d = threadIdx.x; d < D_DIM; d += blockDim.x) {
    atomicAdd(&tdst[d], tsrc[d]);
    atomicAdd(&cdst[d], csrc[d]);
  }
  if (threadIdx.x == 0) atomicAdd(&cnt[seg], 1.0f);
}

// ---------------------------------------------------------------------------
// Normalize segment sums and convert to bf16 GEMM operands.
// ---------------------------------------------------------------------------
__global__ void prep_state_kernel(const float* __restrict__ ts,
                                  const float* __restrict__ cs,
                                  const float* __restrict__ cnt,
                                  __bf16* __restrict__ ts_bf,
                                  __bf16* __restrict__ cs_bf) {
  const int m = blockIdx.x;
  const float inv = 1.0f / cnt[m];
  const size_t base = (size_t)m * D_DIM;
  for (int d = threadIdx.x; d < D_DIM; d += blockDim.x) {
    ts_bf[base + d] = (__bf16)(ts[base + d] * inv);
    cs_bf[base + d] = (__bf16)(cs[base + d] * inv);
  }
}

// ---------------------------------------------------------------------------
// H[r][k] = bf16( tanh( state[r/len][k] + pos[r%len][k] + emb[ids[r]][k] ) )
// ---------------------------------------------------------------------------
__global__ void prep_h_kernel(const float* __restrict__ state,
                              const float* __restrict__ pos,
                              const float* __restrict__ emb,
                              const int* __restrict__ ids, int len,
                              __bf16* __restrict__ H) {
  const int r = blockIdx.x;
  const int t = r / len;
  const int f = r % len;
  const float* s = state + (size_t)t * D_DIM;
  const float* p = pos + (size_t)f * D_DIM;
  const float* e = emb + (size_t)ids[r] * D_DIM;
  __bf16* dst = H + (size_t)r * D_DIM;
  for (int d = threadIdx.x; d < D_DIM; d += blockDim.x)
    dst[d] = (__bf16)tanhf(s[d] + p[d] + e[d]);
}

// ---------------------------------------------------------------------------
// Host-side orchestration.
// ---------------------------------------------------------------------------
extern "C" void kernel_launch(void* const* d_in, const int* in_sizes, int n_in,
                              void* d_out, int out_size, void* d_ws, size_t ws_size,
                              hipStream_t stream) {
  const int*   xtokens  = (const int*)d_in[0];
  const int*   tokens   = (const int*)d_in[1];
  const int*   tfc      = (const int*)d_in[5];
  const int*   ttags    = (const int*)d_in[6];
  const float* xemb     = (const float*)d_in[7];
  const float* W_enc    = (const float*)d_in[8];
  const float* b_enc    = (const float*)d_in[9];
  const float* char_emb = (const float*)d_in[10];
  const float* tag_emb  = (const float*)d_in[11];
  const float* W_s      = (const float*)d_in[12];
  const float* W_c      = (const float*)d_in[13];
  const float* form_pos = (const float*)d_in[14];
  const float* tag_pos  = (const float*)d_in[15];
  const float* W_char   = (const float*)d_in[16];
  const float* W_tag    = (const float*)d_in[17];

  float* out_form = (float*)d_out;                              // (T*F, CV)
  float* out_tag  = out_form + (size_t)T_SEG * F_LEN * CV_DIM;  // (T*L, TV)

  // Workspace layout (bytes). H_form/H_tag overlay dead earlier regions.
  char* ws = (char*)d_ws;
  float*  token_ctx = (float*)(ws + 0);          // 25165824
  __bf16* A_enc     = (__bf16*)(ws + 25165824);  // 12582912
  float*  ts        = (float*)(ws + 37748736);   //  3145728
  float*  cs        = (float*)(ws + 40894464);   //  3145728
  float*  cnt       = (float*)(ws + 44040192);   //     4096
  __bf16* ts_bf     = (__bf16*)(ws + 44044288);  //  1572864
  __bf16* cs_bf     = (__bf16*)(ws + 45617152);  //  1572864
  float*  state     = (float*)(ws + 47190016);   //  3145728
  __bf16* Wt_enc    = (__bf16*)(ws + 50335744);  //  1179648
  __bf16* Wt_s      = (__bf16*)(ws + 51515392);  //  1179648
  __bf16* Wt_c      = (__bf16*)(ws + 52695040);  //  1179648
  __bf16* Wt_char   = (__bf16*)(ws + 53874688);  //   786432
  __bf16* Wt_tag    = (__bf16*)(ws + 54661120);  //   786432
  __bf16* H_form    = (__bf16*)(ws + 0);         // reuse token_ctx
  __bf16* H_tag     = (__bf16*)(ws + 25165824);  // reuse A_enc
  // Peak usage: 55,447,552 bytes.

  // 0) One-time weight transposes to bf16 [N][K].
  transpose_bf16_kernel<<<(D_DIM * D_DIM) / 256, 256, 0, stream>>>(W_enc, Wt_enc, D_DIM, D_DIM);
  transpose_bf16_kernel<<<(D_DIM * D_DIM) / 256, 256, 0, stream>>>(W_s, Wt_s, D_DIM, D_DIM);
  transpose_bf16_kernel<<<(D_DIM * D_DIM) / 256, 256, 0, stream>>>(W_c, Wt_c, D_DIM, D_DIM);
  transpose_bf16_kernel<<<(D_DIM * CV_DIM) / 256, 256, 0, stream>>>(W_char, Wt_char, D_DIM, CV_DIM);
  transpose_bf16_kernel<<<(D_DIM * TV_DIM) / 256, 256, 0, stream>>>(W_tag, Wt_tag, D_DIM, TV_DIM);

  // 1) Gather + convert encoder inputs.
  enc_gather_kernel<<<N_TOK, 256, 0, stream>>>(xemb, xtokens, A_enc);

  // 2) token_ctx = tanh(A_enc @ W_enc + b_enc)   [8192x768 @ 768x768]
  gemm_bf16_kernel<<<dim3(D_DIM / BN, N_TOK / BM), 512, 0, stream>>>(
      A_enc, Wt_enc, D_DIM, nullptr, nullptr, 0, b_enc, 1, token_ctx, N_TOK, D_DIM);

  // 3) Zero segment accumulators (ts|cs|cnt contiguous).
  hipMemsetAsync(ts, 0, (size_t)T_SEG * D_DIM * 4 * 2 + 4096, stream);

  // 4) Segment sums.
  segsum_kernel<<<N_TOK, 256, 0, stream>>>(tokens, token_ctx, char_emb, ts, cs, cnt);

  // 5) Normalize + convert.
  prep_state_kernel<<<T_SEG, 256, 0, stream>>>(ts, cs, cnt, ts_bf, cs_bf);

  // 6) state = tanh(ts@W_s + cs@W_c)   [1024x768, dual-K accumulation]
  gemm_bf16_kernel<<<dim3(D_DIM / BN, T_SEG / BM), 512, 0, stream>>>(
      ts_bf, Wt_s, D_DIM, cs_bf, Wt_c, D_DIM, nullptr, 1, state, T_SEG, D_DIM);

  // 7) Form head.
  prep_h_kernel<<<T_SEG * F_LEN, 256, 0, stream>>>(state, form_pos, char_emb, tfc,
                                                   F_LEN, H_form);
  gemm_bf16_kernel<<<dim3(CV_DIM / BN, (T_SEG * F_LEN) / BM), 512, 0, stream>>>(
      H_form, Wt_char, D_DIM, nullptr, nullptr, 0, nullptr, 0, out_form,
      T_SEG * F_LEN, CV_DIM);

  // 8) Tag head.
  prep_h_kernel<<<T_SEG * L_LEN, 256, 0, stream>>>(state, tag_pos, tag_emb, ttags,
                                                   L_LEN, H_tag);
  gemm_bf16_kernel<<<dim3(TV_DIM / BN, (T_SEG * L_LEN) / BM), 512, 0, stream>>>(
      H_tag, Wt_tag, D_DIM, nullptr, nullptr, 0, nullptr, 0, out_tag,
      T_SEG * L_LEN, TV_DIM);
}
